// LlamaAttention_62586263437590
// MI455X (gfx1250) — matrix operand order repaired and checked by reading, not compiled, not run on hardware
//
#include <hip/hip_runtime.h>
#include <stdint.h>

// ---------------- problem constants ----------------
#define S_LEN  2048
#define DMODEL 2048
#define NH     32
#define NKVH   8
#define HD     64
#define QKVN   3072   // H*HD + 2*KVH*HD

typedef __attribute__((ext_vector_type(16))) __bf16 v16bf;
typedef __attribute__((ext_vector_type(8)))  float  v8f;
typedef int v4i_vec __attribute__((__vector_size__(16)));   // matches builtin param

union Frag32B { uint4 u[2]; v16bf v; };   // one 16x32 bf16 WMMA operand (8 VGPRs)

// ---- gfx1250 async-to-LDS path (ASYNCcnt), if the toolchain exposes it ----
#if defined(__has_builtin)
#if __has_builtin(__builtin_amdgcn_global_load_async_to_lds_b128) && \
    __has_builtin(__builtin_amdgcn_s_wait_asynccnt)
#define USE_ASYNC_LDS 1
#endif
#endif
#ifndef USE_ASYNC_LDS
#define USE_ASYNC_LDS 0
#endif

__device__ __forceinline__ unsigned short f2bf(float f) {
  union { float f; unsigned u; } c; c.f = f;
  unsigned u = c.u;
  u += 0x7fffu + ((u >> 16) & 1u);   // RNE
  return (unsigned short)(u >> 16);
}

__device__ __forceinline__ v8f bwmma(const Frag32B& a, const Frag32B& b, v8f c) {
  return __builtin_amdgcn_wmma_f32_16x16x32_bf16(false, a.v, false, b.v,
                                                 (short)0, c, false, false);
}

#if USE_ASYNC_LDS
__device__ __forceinline__ void stage128_async(const uint4* g, uint4* l) {
  __builtin_amdgcn_global_load_async_to_lds_b128(
      (__attribute__((address_space(1))) v4i_vec*)(void*)g,
      (__attribute__((address_space(3))) v4i_vec*)(void*)l, 0, 0);
}
#endif

// ---------------- elementwise conversion / packing ----------------
__global__ void cvt_f32_bf16(const float* __restrict__ src,
                             unsigned short* __restrict__ dst, int n) {
  for (int i = blockIdx.x * blockDim.x + threadIdx.x; i < n;
       i += gridDim.x * blockDim.x)
    dst[i] = f2bf(src[i]);
}

// WqkvT[c][r] = [Wq|Wk|Wv][r][c]  -> [3072][2048] bf16 (B^T for the QKV GEMM)
__global__ void pack_wqkvT(const float* __restrict__ Wq, const float* __restrict__ Wk,
                           const float* __restrict__ Wv, unsigned short* __restrict__ dst) {
  const int n = DMODEL * QKVN;
  for (int i = blockIdx.x * blockDim.x + threadIdx.x; i < n;
       i += gridDim.x * blockDim.x) {
    int c = i / DMODEL, r = i % DMODEL;   // c: output col (N), r: reduction (K)
    float v = (c < 2048) ? Wq[(size_t)r * 2048 + c]
            : (c < 2560) ? Wk[(size_t)r * 512 + (c - 2048)]
                         : Wv[(size_t)r * 512 + (c - 2560)];
    dst[i] = f2bf(v);
  }
}

// dst[n][k] = src[k][n]  (src is Kdim x Ndim f32) -> bf16 B^T
__global__ void transpose_bf16(const float* __restrict__ src,
                               unsigned short* __restrict__ dst, int Ndim, int Kdim) {
  const int n = Ndim * Kdim;
  for (int i = blockIdx.x * blockDim.x + threadIdx.x; i < n;
       i += gridDim.x * blockDim.x) {
    int nn = i / Kdim, kk = i % Kdim;
    dst[i] = f2bf(src[(size_t)kk * Ndim + nn]);
  }
}

// ---------------- bf16 WMMA GEMM: C(f32) = A[M][K] * BT[N][K]^T ----------------
// block = 256 threads (8 waves), tile 128(M) x 64(N), K-step 32, double-buffered LDS.
// wave w: msub=(w&3)*32, nsub=(w>>2)*32 -> 2x2 accumulators, 4 WMMA per K-step.
__global__ __launch_bounds__(256)
void gemm_bf16_bt(const unsigned short* __restrict__ A,
                  const unsigned short* __restrict__ BT,
                  float* __restrict__ C, int M, int N, int K) {
  __shared__ __align__(16) uint4 As[2][512];   // 128x32 bf16 per buffer
  __shared__ __align__(16) uint4 Bs[2][256];   // 64x32 bf16 per buffer

  const int t    = threadIdx.x;
  const int l    = t & 31;
  const int w    = t >> 5;
  const int m0   = blockIdx.y * 128;
  const int n0   = blockIdx.x * 64;
  const int msub = (w & 3) * 32;
  const int nsub = (w >> 2) * 32;
  const int half = (l < 16) ? 0 : 1;   // A K-offset 0/16, B K-offset 0/16
  const int lm   = l & 15;

  const uint4* Ag = reinterpret_cast<const uint4*>(A);
  const uint4* Bg = reinterpret_cast<const uint4*>(BT);

  v8f acc[2][2];
#pragma unroll
  for (int i = 0; i < 2; ++i)
#pragma unroll
    for (int j = 0; j < 2; ++j)
#pragma unroll
      for (int r = 0; r < 8; ++r) acc[i][j][r] = 0.f;

  // staging map (tile-local uint4 index -> row/seg of a [rows][32] bf16 tile)
  const int arow = t >> 2, aseg = t & 3;     // A: idx t -> row 0..63; idx t+256 -> +64
  const int nt = K / 32;

  // fragment read offsets (uint4 units inside a tile)
  const int ab0 = (msub + lm) * 4 + half * 2;
  const int ab1 = (msub + 16 + lm) * 4 + half * 2;
  const int bb0 = (nsub + lm) * 4 + half * 2;
  const int bb1 = (nsub + 16 + lm) * 4 + half * 2;

  auto compute = [&](const uint4* as, const uint4* bs) {
    Frag32B a0, a1, b0, b1;
    a0.u[0] = as[ab0]; a0.u[1] = as[ab0 + 1];
    a1.u[0] = as[ab1]; a1.u[1] = as[ab1 + 1];
    b0.u[0] = bs[bb0]; b0.u[1] = bs[bb0 + 1];
    b1.u[0] = bs[bb1]; b1.u[1] = bs[bb1 + 1];
    acc[0][0] = bwmma(a0, b0, acc[0][0]);
    acc[0][1] = bwmma(a0, b1, acc[0][1]);
    acc[1][0] = bwmma(a1, b0, acc[1][0]);
    acc[1][1] = bwmma(a1, b1, acc[1][1]);
  };

#if USE_ASYNC_LDS
  // ---- async DMA pipeline: next tile flows mem->LDS while WMMAs run ----
  auto issue = [&](int kt, int buf) {
    stage128_async(&Ag[(((size_t)(m0 + arow)) * K + kt + aseg * 8) >> 3], &As[buf][t]);
    stage128_async(&Ag[(((size_t)(m0 + 64 + arow)) * K + kt + aseg * 8) >> 3], &As[buf][t + 256]);
    stage128_async(&Bg[(((size_t)(n0 + arow)) * K + kt + aseg * 8) >> 3], &Bs[buf][t]);
  };
  issue(0, 0);
  for (int i = 0; i < nt; ++i) {
    if (i + 1 < nt) {
      issue((i + 1) * 32, (i + 1) & 1);
      __builtin_amdgcn_s_wait_asynccnt(3);   // tile i landed (in-order completion)
    } else {
      __builtin_amdgcn_s_wait_asynccnt(0);
    }
    __syncthreads();
    compute(As[i & 1], Bs[i & 1]);
    __syncthreads();
  }
#else
  // ---- register software pipeline: issue next-tile loads, then compute ----
  uint4 ra0, ra1, rb;
  auto fetch = [&](int kt) {
    ra0 = Ag[(((size_t)(m0 + arow)) * K + kt + aseg * 8) >> 3];
    ra1 = Ag[(((size_t)(m0 + 64 + arow)) * K + kt + aseg * 8) >> 3];
    rb  = Bg[(((size_t)(n0 + arow)) * K + kt + aseg * 8) >> 3];
  };
  fetch(0);
  for (int i = 0; i < nt; ++i) {
    As[i & 1][t]       = ra0;
    As[i & 1][t + 256] = ra1;
    Bs[i & 1][t]       = rb;
    __syncthreads();
    if (i + 1 < nt) fetch((i + 1) * 32);     // loads in flight during WMMAs
    compute(As[i & 1], Bs[i & 1]);
    __syncthreads();
  }
#endif

  // epilogue: C/D layout row m = r + half*8, col n = lane&15
#pragma unroll
  for (int si = 0; si < 2; ++si)
#pragma unroll
    for (int sj = 0; sj < 2; ++sj)
#pragma unroll
      for (int r = 0; r < 8; ++r) {
        int row = m0 + msub + si * 16 + r + half * 8;
        int col = n0 + nsub + sj * 16 + lm;
        C[(size_t)row * N + col] = acc[si][sj][r];
      }
}

// ---------------- RoPE + split + layout change ----------------
__global__ __launch_bounds__(256)
void rope_split(const float* __restrict__ QKV,
                unsigned short* __restrict__ Qb,
                unsigned short* __restrict__ Kb,
                unsigned short* __restrict__ Vt) {
  const int s = blockIdx.x;
  const int t = threadIdx.x;
  const float* row = QKV + (size_t)s * QKVN;

  for (int i = t; i < NH * 32; i += 256) {          // Q: 32 heads x 32 pairs
    int hh = i >> 5, p = i & 31;
    float xr = row[hh * 64 + 2 * p];
    float xi = row[hh * 64 + 2 * p + 1];
    float inv = __powf(10000.f, -(float)p / 32.f);
    float ang = (float)s * inv;
    float sn, cs; __sincosf(ang, &sn, &cs);
    size_t o = (((size_t)hh * S_LEN) + s) * HD + 2 * p;
    Qb[o]     = f2bf((xr * cs - xi * sn) * 0.125f); // fold 1/sqrt(HD), exact
    Qb[o + 1] = f2bf((xr * sn + xi * cs) * 0.125f);
  }
  for (int i = t; i < NKVH * 32; i += 256) {        // K: 8 kv-heads x 32 pairs
    int hh = i >> 5, p = i & 31;
    float xr = row[2048 + hh * 64 + 2 * p];
    float xi = row[2048 + hh * 64 + 2 * p + 1];
    float inv = __powf(10000.f, -(float)p / 32.f);
    float ang = (float)s * inv;
    float sn, cs; __sincosf(ang, &sn, &cs);
    size_t o = (((size_t)hh * S_LEN) + s) * HD + 2 * p;
    Kb[o]     = f2bf(xr * cs - xi * sn);
    Kb[o + 1] = f2bf(xr * sn + xi * cs);
  }
  for (int i = t; i < NKVH * HD; i += 256) {        // V transpose -> [kvh][64][s]
    int hh = i >> 6, d = i & 63;
    Vt[(((size_t)hh * HD) + d) * S_LEN + s] = f2bf(row[2560 + i]);
  }
}

// ---------------- causal flash attention (bf16 WMMA) ----------------
__global__ __launch_bounds__(128)
void attn_wmma(const unsigned short* __restrict__ Qb,
               const unsigned short* __restrict__ Kb,
               const unsigned short* __restrict__ Vt,
               unsigned short* __restrict__ Ob) {
  __shared__ __align__(16) unsigned short Pt[4][16 * 32]; // per-wave P tile

  const int t    = threadIdx.x;
  const int l    = t & 31;
  const int w    = t >> 5;
  const int h    = blockIdx.y;
  const int kv   = h >> 2;
  const int q0   = blockIdx.x * 64 + w * 16;
  const int half = (l < 16) ? 0 : 1;
  const int lm   = l & 15;
  const int kbA  = half * 16;
  const int kbB  = half * 16;

  const uint4* Qg = reinterpret_cast<const uint4*>(Qb);
  const uint4* Kg = reinterpret_cast<const uint4*>(Kb);
  const uint4* Vg = reinterpret_cast<const uint4*>(Vt);

  Frag32B qf[2];
  {
    size_t rowbase = (((size_t)h * S_LEN) + (q0 + lm)) * HD;
#pragma unroll
    for (int c = 0; c < 2; ++c) {
      size_t e = rowbase + c * 32 + kbA;
      qf[c].u[0] = Qg[e >> 3];
      qf[c].u[1] = Qg[(e + 8) >> 3];
    }
  }

  float mrow[8], lrow[8], alpha[8];
  v8f o[4];
#pragma unroll
  for (int r = 0; r < 8; ++r) { mrow[r] = -1e30f; lrow[r] = 0.f; }
#pragma unroll
  for (int j = 0; j < 4; ++j)
#pragma unroll
    for (int r = 0; r < 8; ++r) o[j][r] = 0.f;

  const int kend = q0 + 16;
  for (int k0 = 0; k0 < kend; k0 += 32) {
    v8f st0, st1;
#pragma unroll
    for (int r = 0; r < 8; ++r) { st0[r] = 0.f; st1[r] = 0.f; }
#pragma unroll
    for (int c = 0; c < 2; ++c) {
      Frag32B kf0, kf1;
      size_t b0 = (((size_t)kv * S_LEN) + (k0 + lm)) * HD + c * 32 + kbB;
      size_t b1 = (((size_t)kv * S_LEN) + (k0 + 16 + lm)) * HD + c * 32 + kbB;
      kf0.u[0] = Kg[b0 >> 3]; kf0.u[1] = Kg[(b0 + 8) >> 3];
      kf1.u[0] = Kg[b1 >> 3]; kf1.u[1] = Kg[(b1 + 8) >> 3];
      st0 = bwmma(qf[c], kf0, st0);
      st1 = bwmma(qf[c], kf1, st1);
    }

#pragma unroll
    for (int r = 0; r < 8; ++r) {
      int qrow = q0 + r + half * 8;
      st0[r] = (k0 + lm      > qrow) ? -1e30f : st0[r];
      st1[r] = (k0 + 16 + lm > qrow) ? -1e30f : st1[r];
    }
#pragma unroll
    for (int r = 0; r < 8; ++r) {
      float v = fmaxf(st0[r], st1[r]);
#pragma unroll
      for (int off = 1; off < 16; off <<= 1)
        v = fmaxf(v, __shfl_xor(v, off, 32));
      float mn = fmaxf(mrow[r], v);
      alpha[r] = __expf(mrow[r] - mn);
      float p0 = __expf(st0[r] - mn);
      float p1 = __expf(st1[r] - mn);
      st0[r] = p0; st1[r] = p1;
      float rs = p0 + p1;
#pragma unroll
      for (int off = 1; off < 16; off <<= 1)
        rs += __shfl_xor(rs, off, 32);
      lrow[r] = alpha[r] * lrow[r] + rs;
      mrow[r] = mn;
    }

    unsigned short* pw = Pt[w];
#pragma unroll
    for (int r = 0; r < 8; ++r) {
      int m = r + half * 8;
      pw[m * 32 + lm]      = f2bf(st0[r]);
      pw[m * 32 + 16 + lm] = f2bf(st1[r]);
    }
    __builtin_amdgcn_wave_barrier();
    asm volatile("s_wait_dscnt 0" ::: "memory");

    Frag32B pf;
    {
      const uint4* P4 = reinterpret_cast<const uint4*>(pw);
      int base = lm * 4 + half * 2;
      pf.u[0] = P4[base];
      pf.u[1] = P4[base + 1];
    }

#pragma unroll
    for (int j = 0; j < 4; ++j) {
#pragma unroll
      for (int r = 0; r < 8; ++r) o[j][r] *= alpha[r];
      Frag32B vf;
      size_t vb = (((size_t)kv * HD) + (j * 16 + lm)) * S_LEN + k0 + kbB;
      vf.u[0] = Vg[vb >> 3]; vf.u[1] = Vg[(vb + 8) >> 3];
      o[j] = bwmma(pf, vf, o[j]);
    }
    __builtin_amdgcn_wave_barrier();
  }

#pragma unroll
  for (int j = 0; j < 4; ++j)
#pragma unroll
    for (int r = 0; r < 8; ++r) {
      int row = q0 + r + half * 8;
      int col = h * HD + j * 16 + lm;
      Ob[(size_t)row * DMODEL + col] = f2bf(o[j][r] / lrow[r]);
    }
}

// ---------------- launcher ----------------
extern "C" void kernel_launch(void* const* d_in, const int* in_sizes, int n_in,
                              void* d_out, int out_size, void* d_ws, size_t ws_size,
                              hipStream_t stream) {
  (void)in_sizes; (void)n_in; (void)out_size; (void)ws_size;
  const float* X  = (const float*)d_in[0];
  const float* Wq = (const float*)d_in[1];
  const float* Wk = (const float*)d_in[2];
  const float* Wv = (const float*)d_in[3];
  const float* Wo = (const float*)d_in[4];
  float* out = (float*)d_out;

  char* ws = (char*)d_ws;
  size_t off = 0;
  auto take = [&](size_t bytes) {
    char* p = ws + off;
    off += (bytes + 255) & ~(size_t)255;
    return p;
  };
  unsigned short* Xb    = (unsigned short*)take((size_t)S_LEN * DMODEL * 2);
  unsigned short* WqkvT = (unsigned short*)take((size_t)DMODEL * QKVN * 2);
  unsigned short* WoT   = (unsigned short*)take((size_t)DMODEL * DMODEL * 2);
  float*          QKV   = (float*)take((size_t)S_LEN * QKVN * 4);
  unsigned short* Qb    = (unsigned short*)take((size_t)NH * S_LEN * HD * 2);
  unsigned short* Kb    = (unsigned short*)take((size_t)NKVH * S_LEN * HD * 2);
  unsigned short* Vt    = (unsigned short*)take((size_t)NKVH * S_LEN * HD * 2);
  unsigned short* Ob    = (unsigned short*)take((size_t)S_LEN * DMODEL * 2);

  cvt_f32_bf16 <<<2048, 256, 0, stream>>>(X, Xb, S_LEN * DMODEL);
  pack_wqkvT   <<<2048, 256, 0, stream>>>(Wq, Wk, Wv, WqkvT);
  transpose_bf16<<<2048, 256, 0, stream>>>(Wo, WoT, DMODEL, DMODEL);

  // QKV = X @ [Wq|Wk|Wv]   (2048 x 3072), B pre-transposed
  gemm_bf16_bt<<<dim3(QKVN / 64, S_LEN / 128), 256, 0, stream>>>(
      Xb, WqkvT, QKV, S_LEN, QKVN, DMODEL);

  rope_split<<<S_LEN, 256, 0, stream>>>(QKV, Qb, Kb, Vt);

  attn_wmma<<<dim3(S_LEN / 64, NH), 128, 0, stream>>>(Qb, Kb, Vt, Ob);

  // out = O @ Wo   (2048 x 2048), B pre-transposed
  gemm_bf16_bt<<<dim3(DMODEL / 64, S_LEN / 128), 256, 0, stream>>>(
      Ob, WoT, out, S_LEN, DMODEL, DMODEL);
}
